// Local_Aggregation_52364241273496
// MI455X (gfx1250) — matrix operand order repaired and checked by reading, not compiled, not run on hardware
//
#include <hip/hip_runtime.h>
#include <hip/hip_bf16.h>

#define CC 128        // channels (in == out)
#define TILE_R 128    // rows (points) per workgroup tile
#define PAD 136       // LDS row pad (bf16 elems): 272B rows -> bank stride 68%64=4
#define EPSV 1e-5f
#define SLOPE 0.01f

typedef __bf16 bf16_t;
typedef __attribute__((ext_vector_type(4)))  float  v4f;
typedef __attribute__((ext_vector_type(8)))  float  v8f;
typedef __attribute__((ext_vector_type(4)))  __bf16 v4bf;
typedef __attribute__((ext_vector_type(8)))  __bf16 v8bf;
typedef __attribute__((ext_vector_type(16))) __bf16 v16bf;

static __device__ __forceinline__ v8f v8f_zero() {
  v8f z;
#pragma unroll
  for (int i = 0; i < 8; ++i) z[i] = 0.0f;
  return z;
}

// D = A(16x32) * B(32x16) + C, bf16 in / f32 out
static __device__ __forceinline__ v8f wmma_bf(v16bf a, v16bf b, v8f c) {
  return __builtin_amdgcn_wmma_f32_16x16x32_bf16(false, a, false, b, (short)0, c,
                                                 false, false);
}

// Async copy 16B global -> LDS (ASYNCcnt-tracked; GV addressing mode).
// Generic shared-pointer low 32 bits == wave-relative LDS byte address.
static __device__ __forceinline__ void async_ld_b128(const void* g, void* l) {
  const unsigned loff = (unsigned)(size_t)l;
  asm volatile("global_load_async_to_lds_b128 %0, %1, off"
               :
               : "v"(loff), "v"(g)
               : "memory");
}
static __device__ __forceinline__ void wait_async0() {
  asm volatile("s_wait_asynccnt 0" ::: "memory");
}

// ---- Fragment loaders -------------------------------------------------------
// A-matrix 16x32 bf16 (ISA 7.12.2): lane L row M=L&15; hi=L>>4 selects K runs
// {8hi..8hi+7} and {16+8hi..+7}. Row-major source: two contiguous 16B chunks.
static __device__ __forceinline__ v16bf
afrag_bf16(const bf16_t* base, int ld, int row0, int k0, int l15, int hi) {
  const bf16_t* r = base + (size_t)(row0 + l15) * ld + k0 + hi * 8;
  v8bf lo = *(const v8bf*)(r);
  v8bf hh = *(const v8bf*)(r + 16);
  v16bf a;
#pragma unroll
  for (int i = 0; i < 8; ++i) { a[i] = lo[i]; a[8 + i] = hh[i]; }
  return a;
}

static __device__ __forceinline__ v16bf
afrag_f32(const float* base, int ld, int row0, int k0, int l15, int hi) {
  const float* r = base + (size_t)(row0 + l15) * ld + k0 + hi * 8;
  v16bf a;
#pragma unroll
  for (int i = 0; i < 8; ++i) { a[i] = (bf16_t)r[i]; a[8 + i] = (bf16_t)r[16 + i]; }
  return a;
}

// B-matrix 32x16 bf16: lane L col N=L&15, K runs {8hi..8hi+7} and {16+8hi..+7}.
// Source layout: row-major [N][K] (e.g. weights stored [out_ch][in_ch]).
static __device__ __forceinline__ v16bf
bfrag_bf16(const bf16_t* base, int ld, int n0, int k0, int l15, int hi) {
  const bf16_t* r = base + (size_t)(n0 + l15) * ld + k0 + hi * 8;
  v8bf lo = *(const v8bf*)(r);
  v8bf hh = *(const v8bf*)(r + 16);
  v16bf b;
#pragma unroll
  for (int i = 0; i < 8; ++i) { b[i] = lo[i]; b[8 + i] = hh[i]; }
  return b;
}

// ---- Kernel 0: zero the stat accumulators ----------------------------------
__global__ void k_zero(float* p, int n) {
  int i = blockIdx.x * 256 + threadIdx.x;
  if (i < n) p[i] = 0.0f;
}

// ---- Kernel 1: per-channel sum(x) + Gram = X^T X (WMMA), optional bf16 copy -
template <bool WRITE_BF16>
__global__ void __launch_bounds__(256)
k_stats_x(const float* __restrict__ x, bf16_t* __restrict__ xbf,
          float* __restrict__ sumx, float* __restrict__ gram, int ntiles) {
  __shared__ bf16_t lds_xt[CC * PAD];   // transposed tile: [channel][local_row]
  __shared__ float  lds_sum[CC];
  const int t = threadIdx.x;
  const int wv = t >> 5, lane = t & 31, l15 = lane & 15, hi = lane >> 4;
  const int c0 = (t * 4) & (CC - 1);

  float ls0 = 0.f, ls1 = 0.f, ls2 = 0.f, ls3 = 0.f;
  v8f gacc[8];
#pragma unroll
  for (int j = 0; j < 8; ++j) gacc[j] = v8f_zero();

  for (int tile = blockIdx.x; tile < ntiles; tile += gridDim.x) {
    const size_t base = (size_t)tile * TILE_R * CC;
#pragma unroll
    for (int e = 0; e < 16; ++e) {
      const size_t flat = base + (size_t)e * 1024 + (size_t)t * 4;
      v4f f = *(const v4f*)(x + flat);
      v4bf bq;
      bq[0] = (bf16_t)f[0]; bq[1] = (bf16_t)f[1];
      bq[2] = (bf16_t)f[2]; bq[3] = (bf16_t)f[3];
      if (WRITE_BF16) *(v4bf*)(xbf + flat) = bq;
      const int rowl = 8 * e + wv;  // (e*1024 + 4t)/128
      lds_xt[(c0 + 0) * PAD + rowl] = bq[0];
      lds_xt[(c0 + 1) * PAD + rowl] = bq[1];
      lds_xt[(c0 + 2) * PAD + rowl] = bq[2];
      lds_xt[(c0 + 3) * PAD + rowl] = bq[3];
      ls0 += f[0]; ls1 += f[1]; ls2 += f[2]; ls3 += f[3];
    }
    __syncthreads();
    // Gram: G[i,j] += sum_p xt[i][p]*xt[j][p]; wave owns i-slab 16*wv
#pragma unroll
    for (int kt = 0; kt < 4; ++kt) {
      v16bf a = afrag_bf16(lds_xt, PAD, 16 * wv, 32 * kt, l15, hi);
#pragma unroll
      for (int jt = 0; jt < 8; ++jt) {
        v16bf b = bfrag_bf16(lds_xt, PAD, 16 * jt, 32 * kt, l15, hi);
        gacc[jt] = wmma_bf(a, b, gacc[jt]);
      }
    }
    __syncthreads();
  }

  if (t < CC) lds_sum[t] = 0.f;
  __syncthreads();
  atomicAdd(&lds_sum[c0 + 0], ls0);
  atomicAdd(&lds_sum[c0 + 1], ls1);
  atomicAdd(&lds_sum[c0 + 2], ls2);
  atomicAdd(&lds_sum[c0 + 3], ls3);
  __syncthreads();
  if (t < CC) atomicAdd(&sumx[t], lds_sum[t]);
#pragma unroll
  for (int jt = 0; jt < 8; ++jt)
#pragma unroll
    for (int r = 0; r < 8; ++r) {
      const int i = 16 * wv + r + 8 * hi;
      const int j = 16 * jt + l15;
      atomicAdd(&gram[i * CC + j], gacc[jt][r]);
    }
}

// ---- Kernel 2: fold BN stats into effective bf16 weights + fp32 shift ------
// mean(Wx+b)=W.mu+b ; var(Wx)=W Cov W^T ; w_eff=W*S ; b_eff=beta-(W.mu)*S
__global__ void __launch_bounds__(128)
k_finalize(const float* __restrict__ w, const float* __restrict__ bias,
           const float* __restrict__ g, const float* __restrict__ be,
           const float* __restrict__ sums, const float* __restrict__ gram,
           float invN, bf16_t* __restrict__ we, float* __restrict__ beff) {
  __shared__ float mu[CC];
  const int o = threadIdx.x;
  mu[o] = sums[o] * invN;
  __syncthreads();
  float wm = 0.f;
  for (int i = 0; i < CC; ++i) wm += w[o * CC + i] * mu[i];
  float e2 = 0.f;
  for (int i = 0; i < CC; ++i) {
    float acc = 0.f;
    for (int j = 0; j < CC; ++j) acc += gram[i * CC + j] * w[o * CC + j];
    e2 += w[o * CC + i] * acc;
  }
  const float var = e2 * invN - wm * wm;
  const float S = g[o] * rsqrtf(var + EPSV);
  for (int i = 0; i < CC; ++i) we[o * CC + i] = (bf16_t)(w[o * CC + i] * S);
  beff[o] = be[o] - wm * S;
  (void)bias;  // bias cancels out of the folded shift; var unaffected by it
}

// ---- Kernel 3: y1 = leaky(X.W1e^T + b1e); accumulate sum(y1), Gram(y1) -----
template <bool XBF>
__global__ void __launch_bounds__(256)
k_stats_y(const void* __restrict__ xsrc, const bf16_t* __restrict__ w1e,
          const float* __restrict__ b1e, float* __restrict__ sumy,
          float* __restrict__ gramy, int ntiles) {
  __shared__ bf16_t lds_x[TILE_R * PAD];  // async-staged X tile (bf16 path)
  __shared__ bf16_t lds_yt[CC * PAD];     // transposed y1 tile
  const int t = threadIdx.x;
  const int wv = t >> 5, lane = t & 31, l15 = lane & 15, hi = lane >> 4;

  v8f gacc[8];
  float csum[8], bias1[8];
#pragma unroll
  for (int n = 0; n < 8; ++n) {
    gacc[n] = v8f_zero();
    csum[n] = 0.f;
    bias1[n] = b1e[16 * n + l15];
  }

  for (int tile = blockIdx.x; tile < ntiles; tile += gridDim.x) {
    const int row0 = tile * TILE_R + 16 * wv;
    if (XBF) {
      // Coalesced async stage: 16 lanes cover one 256B row, 16B per lane.
      const bf16_t* xb = (const bf16_t*)xsrc;
#pragma unroll
      for (int e = 0; e < 8; ++e) {
        const int chunk = e * 256 + t;        // 2048 x 16B chunks
        const int row = chunk >> 4;
        const int c8 = (chunk & 15) * 8;
        async_ld_b128(xb + (size_t)(tile * TILE_R + row) * CC + c8,
                      &lds_x[row * PAD + c8]);
      }
      wait_async0();
    }
    __syncthreads();

    v8f acc[8];
#pragma unroll
    for (int n = 0; n < 8; ++n) acc[n] = v8f_zero();
#pragma unroll
    for (int kt = 0; kt < 4; ++kt) {
      v16bf a = XBF ? afrag_bf16(lds_x, PAD, 16 * wv, 32 * kt, l15, hi)
                    : afrag_f32((const float*)xsrc, CC, row0, 32 * kt, l15, hi);
#pragma unroll
      for (int n = 0; n < 8; ++n) {
        v16bf b = bfrag_bf16(w1e, CC, 16 * n, 32 * kt, l15, hi);
        acc[n] = wmma_bf(a, b, acc[n]);
      }
    }
#pragma unroll
    for (int n = 0; n < 8; ++n)
#pragma unroll
      for (int r = 0; r < 8; ++r) {
        float v = acc[n][r] + bias1[n];
        v = v > 0.f ? v : SLOPE * v;
        csum[n] += v;
        lds_yt[(16 * n + l15) * PAD + (16 * wv + r + 8 * hi)] = (bf16_t)v;
      }
    __syncthreads();
#pragma unroll
    for (int kt = 0; kt < 4; ++kt) {
      v16bf a = afrag_bf16(lds_yt, PAD, 16 * wv, 32 * kt, l15, hi);
#pragma unroll
      for (int jt = 0; jt < 8; ++jt) {
        v16bf b = bfrag_bf16(lds_yt, PAD, 16 * jt, 32 * kt, l15, hi);
        gacc[jt] = wmma_bf(a, b, gacc[jt]);
      }
    }
    __syncthreads();
  }

#pragma unroll
  for (int n = 0; n < 8; ++n) atomicAdd(&sumy[16 * n + l15], csum[n]);
#pragma unroll
  for (int jt = 0; jt < 8; ++jt)
#pragma unroll
    for (int r = 0; r < 8; ++r) {
      const int i = 16 * wv + r + 8 * hi;
      const int j = 16 * jt + l15;
      atomicAdd(&gramy[i * CC + j], gacc[jt][r]);
    }
}

// ---- Kernel 4: fused GEMM1 -> leaky -> GEMM2 -> +residual -> leaky -> max --
template <bool XBF>
__global__ void __launch_bounds__(256)
k_final(const void* __restrict__ xsrc, const bf16_t* __restrict__ w1e,
        const float* __restrict__ b1e, const bf16_t* __restrict__ w2e,
        const float* __restrict__ b2e, float* __restrict__ out, int ntiles) {
  __shared__ bf16_t lds_x[TILE_R * PAD];  // async-staged X tile (bf16 path)
  __shared__ bf16_t lds_y[TILE_R * PAD];  // y1 tile, row-major [p][ch]
  __shared__ float  pmax[8 * CC];
  const int t = threadIdx.x;
  const int wv = t >> 5, lane = t & 31, l15 = lane & 15, hi = lane >> 4;
  const int tile = blockIdx.x;
  const int row0 = tile * TILE_R + 16 * wv;

  if (XBF) {
    const bf16_t* xb = (const bf16_t*)xsrc;
#pragma unroll
    for (int e = 0; e < 8; ++e) {
      const int chunk = e * 256 + t;
      const int row = chunk >> 4;
      const int c8 = (chunk & 15) * 8;
      async_ld_b128(xb + (size_t)(tile * TILE_R + row) * CC + c8,
                    &lds_x[row * PAD + c8]);
    }
    wait_async0();
  }
  __syncthreads();

  // GEMM1 + BN1(folded) + leaky -> LDS (row-major for GEMM2 A-fragments)
  {
    v8f acc[8];
#pragma unroll
    for (int n = 0; n < 8; ++n) acc[n] = v8f_zero();
#pragma unroll
    for (int kt = 0; kt < 4; ++kt) {
      v16bf a = XBF ? afrag_bf16(lds_x, PAD, 16 * wv, 32 * kt, l15, hi)
                    : afrag_f32((const float*)xsrc, CC, row0, 32 * kt, l15, hi);
#pragma unroll
      for (int n = 0; n < 8; ++n) {
        v16bf b = bfrag_bf16(w1e, CC, 16 * n, 32 * kt, l15, hi);
        acc[n] = wmma_bf(a, b, acc[n]);
      }
    }
#pragma unroll
    for (int n = 0; n < 8; ++n) {
      const float bb = b1e[16 * n + l15];
#pragma unroll
      for (int r = 0; r < 8; ++r) {
        float v = acc[n][r] + bb;
        v = v > 0.f ? v : SLOPE * v;
        lds_y[(16 * wv + r + 8 * hi) * PAD + 16 * n + l15] = (bf16_t)v;
      }
    }
  }
  __syncthreads();

  // GEMM2 + BN2(folded) + residual + leaky + partial max over wave's 16 rows
  v8f acc2[8];
#pragma unroll
  for (int n = 0; n < 8; ++n) acc2[n] = v8f_zero();
#pragma unroll
  for (int kt = 0; kt < 4; ++kt) {
    v16bf a = afrag_bf16(lds_y, PAD, 16 * wv, 32 * kt, l15, hi);
#pragma unroll
    for (int n = 0; n < 8; ++n) {
      v16bf b = bfrag_bf16(w2e, CC, 16 * n, 32 * kt, l15, hi);
      acc2[n] = wmma_bf(a, b, acc2[n]);
    }
  }
#pragma unroll
  for (int n = 0; n < 8; ++n) {
    const int c = 16 * n + l15;
    const float bb = b2e[c];
    float m = -3.4e38f;
#pragma unroll
    for (int r = 0; r < 8; ++r) {
      const int rl = 16 * wv + r + 8 * hi;     // local row in tile
      const float res =
          XBF ? (float)lds_x[rl * PAD + c]
              : ((const float*)xsrc)[((size_t)tile * TILE_R + rl) * CC + c];
      float v = acc2[n][r] + bb + res;
      v = v > 0.f ? v : SLOPE * v;
      m = fmaxf(m, v);
    }
    m = fmaxf(m, __shfl_xor(m, 16, 32));
    if (hi == 0) pmax[wv * CC + c] = m;
  }
  __syncthreads();

  // Combine wave-pair partials: group g (32 rows) = waves 2g, 2g+1
  for (int idx = t; idx < 4 * CC; idx += 256) {
    const int g = idx >> 7, c = idx & (CC - 1);
    const float v = fmaxf(pmax[(2 * g) * CC + c], pmax[(2 * g + 1) * CC + c]);
    out[((size_t)tile * 4 + g) * CC + c] = v;
  }
  (void)ntiles;
}

// ---- Host launcher ----------------------------------------------------------
extern "C" void kernel_launch(void* const* d_in, const int* in_sizes, int n_in,
                              void* d_out, int out_size, void* d_ws, size_t ws_size,
                              hipStream_t stream) {
  const float* x   = (const float*)d_in[0];
  const float* w1  = (const float*)d_in[1];
  const float* b1  = (const float*)d_in[2];
  const float* g1  = (const float*)d_in[3];
  const float* be1 = (const float*)d_in[4];
  const float* w2  = (const float*)d_in[5];
  const float* b2  = (const float*)d_in[6];
  const float* g2  = (const float*)d_in[7];
  const float* be2 = (const float*)d_in[8];
  float* out = (float*)d_out;

  const int P = in_sizes[0] / CC;      // total points = b*n*s (524288)
  const int ntiles = P / TILE_R;       // 4096
  const float invN = 1.0f / (float)P;

  char* ws = (char*)d_ws;
  float*  sumx  = (float*)(ws);
  float*  gramx = sumx + CC;
  float*  sumy  = gramx + CC * CC;
  float*  gramy = sumy + CC;
  float*  b1e   = gramy + CC * CC;
  float*  b2e   = b1e + CC;
  bf16_t* w1e   = (bf16_t*)(b2e + CC);
  bf16_t* w2e   = w1e + CC * CC;
  bf16_t* xbf   = (bf16_t*)(ws + 262144);
  const bool use_bf = ws_size >= 262144 + (size_t)P * CC * sizeof(bf16_t);

  const int nstat = 2 * (CC + CC * CC);
  k_zero<<<(nstat + 255) / 256, 256, 0, stream>>>(sumx, nstat);

  if (use_bf) {
    k_stats_x<true><<<256, 256, 0, stream>>>(x, xbf, sumx, gramx, ntiles);
    k_finalize<<<1, 128, 0, stream>>>(w1, b1, g1, be1, sumx, gramx, invN, w1e, b1e);
    k_stats_y<true><<<256, 256, 0, stream>>>(xbf, w1e, b1e, sumy, gramy, ntiles);
    k_finalize<<<1, 128, 0, stream>>>(w2, b2, g2, be2, sumy, gramy, invN, w2e, b2e);
    k_final<true><<<ntiles, 256, 0, stream>>>(xbf, w1e, b1e, w2e, b2e, out, ntiles);
  } else {
    k_stats_x<false><<<256, 256, 0, stream>>>(x, nullptr, sumx, gramx, ntiles);
    k_finalize<<<1, 128, 0, stream>>>(w1, b1, g1, be1, sumx, gramx, invN, w1e, b1e);
    k_stats_y<false><<<256, 256, 0, stream>>>(x, w1e, b1e, sumy, gramy, ntiles);
    k_finalize<<<1, 128, 0, stream>>>(w2, b2, g2, be2, sumy, gramy, invN, w2e, b2e);
    k_final<false><<<ntiles, 256, 0, stream>>>(x, w1e, b1e, w2e, b2e, out, ntiles);
  }
  (void)n_in; (void)out_size;
}
